// Faster_Rcnn_54820962566981
// MI455X (gfx1250) — matrix operand order repaired and checked by reading, not compiled, not run on hardware
//
#include <hip/hip_runtime.h>
#include <hip/hip_bf16.h>

// ---------------------------------------------------------------------------
// MI455X (gfx1250) Faster-RCNN loss pipeline.
// Heavy math (VGG convs, RPN conv, FC1/FC2) runs through
// v_wmma_f32_16x16x32_bf16 (wave32, f32 accumulate).
// Block tile 128x64, 8 waves, 32x32 per wave (4 WMMAs / wave / k-step).
// LDS tiles are laid out so every WMMA fragment is 2x ds_load_b128:
//   A row-major [m][k], 80B row stride  -> lane reads K[8h..8h+7],[16+8h..23+8h]
//   B transposed [n][k], 80B row stride -> lane reads K[8h..8h+7],[16+8h..23+8h]
// Staging is 16-byte granular: conv weights are pre-padded to K%32==0,
// FC weights are pre-transposed to [N][K] so all tile copies are b128.
// ---------------------------------------------------------------------------

typedef __bf16 bfloat;
typedef __attribute__((ext_vector_type(16))) __bf16 v16bf;
typedef __attribute__((ext_vector_type(8)))  __bf16 v8bf;
typedef __attribute__((ext_vector_type(8)))  float  v8f;
typedef __attribute__((ext_vector_type(4)))  float  v4f;
typedef __attribute__((ext_vector_type(4)))  unsigned int v4u;

#define LDA 40  // bf16 elements per LDS row (80 bytes, 16B aligned)

__device__ __forceinline__ unsigned short f2bf_bits(float f) {
  union { float f; unsigned u; } c; c.f = f;
  unsigned r = c.u + 0x7FFFu + ((c.u >> 16) & 1u);  // round-to-nearest-even
  return (unsigned short)(r >> 16);
}
__device__ __forceinline__ bfloat f2bf(float f) {
  unsigned short h = f2bf_bits(f);
  bfloat out;
  __builtin_memcpy(&out, &h, 2);
  return out;
}

// A fragment (16x32 MxK): lane holds row M=lane%16, half=lane/16.
// Elements e=0..7 -> K=8h+e ; e=8..15 -> K=16+8h+(e-8): two 16B runs.
__device__ __forceinline__ v16bf frag_a(const bfloat lA[][LDA], int row, int lane) {
  int mrow = row + (lane & 15);
  int half = lane >> 4;
  v8bf a0 = *(const v8bf*)&lA[mrow][8 * half];
  v8bf a1 = *(const v8bf*)&lA[mrow][16 + 8 * half];
  return __builtin_shufflevector(a0, a1, 0, 1, 2, 3, 4, 5, 6, 7,
                                 8, 9, 10, 11, 12, 13, 14, 15);
}
// B fragment (32x16 KxN): lane holds col N=lane%16, same K order as A.
// With B stored transposed [n][k], that is two 16B runs.
__device__ __forceinline__ v16bf frag_b(const bfloat lBt[][LDA], int col, int lane) {
  int ncol = col + (lane & 15);
  int half = lane >> 4;
  v8bf b0 = *(const v8bf*)&lBt[ncol][8 * half];
  v8bf b1 = *(const v8bf*)&lBt[ncol][16 + 8 * half];
  return __builtin_shufflevector(b0, b1, 0, 1, 2, 3, 4, 5, 6, 7,
                                 8, 9, 10, 11, 12, 13, 14, 15);
}

// ----------------------- small utility kernels -----------------------------
__global__ void zero_kernel(float* p, int n) {
  int i = blockIdx.x * blockDim.x + threadIdx.x;
  if (i < n) p[i] = 0.f;
}

// conv weights: [M][K] f32 -> [M][Kpad] bf16, zero-padded K tail
__global__ void cvt_pad_bf16_kernel(const float* __restrict__ src,
                                    bfloat* __restrict__ dst,
                                    int M, int K, int Kpad) {
  int i = blockIdx.x * blockDim.x + threadIdx.x;
  if (i >= M * Kpad) return;
  int m = i / Kpad, k = i % Kpad;
  dst[i] = (k < K) ? f2bf(src[(size_t)m * K + k]) : f2bf(0.f);
}

// FC weights: [K][N] f32 -> [N][K] bf16 (transpose so GEMM B-staging is b128)
__global__ void cvt_transpose_bf16_kernel(const float* __restrict__ src,
                                          bfloat* __restrict__ dst,
                                          int K, int N) {
  size_t i = (size_t)blockIdx.x * blockDim.x + threadIdx.x;
  if (i >= (size_t)N * K) return;
  size_t n = i / K, k = i % K;
  dst[i] = f2bf(src[k * (size_t)N + n]);
}

// img (H,W,3 RGB) -> x (3,H,W) BGR, mean-subtracted. scale==1 for 600x800.
__global__ void preprocess_kernel(const float* __restrict__ img,
                                  float* __restrict__ x, int H, int W) {
  int i = blockIdx.x * blockDim.x + threadIdx.x;
  int HW = H * W;
  if (i >= 3 * HW) return;
  int c = i / HW, p = i % HW;
  const float MEANB[3] = {122.7717f, 115.9465f, 102.9801f};
  x[i] = img[(size_t)p * 3 + (2 - c)] - MEANB[c];
}

// ----------------------- implicit-GEMM 3x3 conv (WMMA) ---------------------
// out[Cout][H][W] = relu(conv3x3_same(in[Cin][H][W], wts) + bias)
// GEMM view: M=Cout, N=H*W, K=Cin*9 (weights padded to Kpad). Wave: 32x32.
__global__ __launch_bounds__(256) void conv3x3_wmma_kernel(
    const float* __restrict__ in, const bfloat* __restrict__ wts,
    const float* __restrict__ bias, float* __restrict__ out,
    int Cin, int Cout, int H, int W, int relu) {
  const int HW = H * W;
  const int K = Cin * 9;
  const int Kpad = (K + 31) & ~31;
  const int ntn = (HW + 63) / 64;
  const int bn = blockIdx.x % ntn;
  const int bm = blockIdx.x / ntn;
  const int m0 = bm * 128, n0 = bn * 64;

  __shared__ bfloat lA[128][LDA];  // A tile, row-major [m][k]
  __shared__ bfloat lBt[64][LDA];  // B tile, transposed [n][k]

  const int tid = threadIdx.x;
  const int lane = tid & 31;
  const int wave = tid >> 5;
  const int wm = (wave & 3) * 32;   // wave's 32-row slice
  const int wn = (wave >> 2) * 32;  // wave's 32-col slice

  v8f acc00 = {}, acc01 = {}, acc10 = {}, acc11 = {};

  for (int k0 = 0; k0 < Kpad; k0 += 32) {
    // stage A (padded weights): 128x32 bf16 as 16B copies, row guard only
#pragma unroll
    for (int i = tid; i < 512; i += 256) {  // 512 = 128*32/8 vectors
      int m = i >> 2;
      int koff = (i & 3) * 8;
      v4u v = {0u, 0u, 0u, 0u};
      if (m0 + m < Cout)
        v = *(const v4u*)&wts[(size_t)(m0 + m) * Kpad + k0 + koff];
      *(v4u*)&lA[m][koff] = v;
    }
    // stage B (im2col on the fly): 32x64, stored transposed
    for (int i = tid; i < 32 * 64; i += 256) {
      int k = i >> 6, n = i & 63;
      int gk = k0 + k, gp = n0 + n;
      float v = 0.f;
      if (gk < K && gp < HW) {
        int ci = gk / 9, rs = gk % 9;
        int r = rs / 3 - 1, s = rs % 3 - 1;
        int y = gp / W + r, x = gp % W + s;
        if (y >= 0 && y < H && x >= 0 && x < W)
          v = in[(size_t)ci * HW + (size_t)y * W + x];
      }
      lBt[n][k] = f2bf(v);
    }
    __syncthreads();

    v16bf a0 = frag_a(lA, wm, lane);
    v16bf a1 = frag_a(lA, wm + 16, lane);
    v16bf b0 = frag_b(lBt, wn, lane);
    v16bf b1 = frag_b(lBt, wn + 16, lane);
    acc00 = __builtin_amdgcn_wmma_f32_16x16x32_bf16(false, a0, false, b0,
                                                    (short)0, acc00, false, false);
    acc01 = __builtin_amdgcn_wmma_f32_16x16x32_bf16(false, a0, false, b1,
                                                    (short)0, acc01, false, false);
    acc10 = __builtin_amdgcn_wmma_f32_16x16x32_bf16(false, a1, false, b0,
                                                    (short)0, acc10, false, false);
    acc11 = __builtin_amdgcn_wmma_f32_16x16x32_bf16(false, a1, false, b1,
                                                    (short)0, acc11, false, false);
    __syncthreads();
  }

  // C/D layout: VGPR j -> row j + 8*(lane/16); col = lane%16
  const int half = lane >> 4;
  const int ncol = lane & 15;
#pragma unroll
  for (int rm = 0; rm < 2; ++rm) {
#pragma unroll
    for (int rn = 0; rn < 2; ++rn) {
      const v8f* acc = rm == 0 ? (rn == 0 ? &acc00 : &acc01)
                               : (rn == 0 ? &acc10 : &acc11);
#pragma unroll
      for (int j = 0; j < 8; ++j) {
        int m = m0 + wm + rm * 16 + half * 8 + j;
        int p = n0 + wn + rn * 16 + ncol;
        if (m < Cout && p < HW) {
          float v = (*acc)[j] + bias[m];
          if (relu) v = fmaxf(v, 0.f);
          out[(size_t)m * HW + p] = v;
        }
      }
    }
  }
}

// ----------------------- WMMA GEMM (FC layers) -----------------------------
// C[MxN] = relu(A[MxK] (f32) @ B (bf16, pre-transposed [N][K]) + bias[N])
// Requires K % 32 == 0 (holds: 25088, 4096).
__global__ __launch_bounds__(256) void gemm_wmma_kernel(
    const float* __restrict__ A, const bfloat* __restrict__ Bt,
    const float* __restrict__ bias, float* __restrict__ C,
    int M, int N, int K, int relu) {
  const int ntn = (N + 63) / 64;
  const int bn = blockIdx.x % ntn;
  const int bm = blockIdx.x / ntn;
  const int m0 = bm * 128, n0 = bn * 64;

  __shared__ bfloat lA[128][LDA];
  __shared__ bfloat lBt[64][LDA];

  const int tid = threadIdx.x;
  const int lane = tid & 31;
  const int wave = tid >> 5;
  const int wm = (wave & 3) * 32;
  const int wn = (wave >> 2) * 32;

  v8f acc00 = {}, acc01 = {}, acc10 = {}, acc11 = {};

  for (int k0 = 0; k0 < K; k0 += 32) {
    // stage A (f32 activations): 2x v4f b128 loads -> packed bf16 16B store
#pragma unroll
    for (int i = tid; i < 512; i += 256) {
      int m = i >> 2;
      int koff = (i & 3) * 8;
      union { v4u u4; unsigned short s[8]; } o;
      if (m0 + m < M) {
        const float* src = &A[(size_t)(m0 + m) * K + k0 + koff];
        v4f f0 = *(const v4f*)&src[0];
        v4f f1 = *(const v4f*)&src[4];
#pragma unroll
        for (int e = 0; e < 4; ++e) o.s[e] = f2bf_bits(f0[e]);
#pragma unroll
        for (int e = 0; e < 4; ++e) o.s[4 + e] = f2bf_bits(f1[e]);
      } else {
        o.u4 = (v4u){0u, 0u, 0u, 0u};
      }
      *(v4u*)&lA[m][koff] = o.u4;
    }
    // stage B (pre-transposed weights): pure 16B copies
#pragma unroll
    for (int i = tid; i < 256; i += 256) {  // 256 = 64*32/8 vectors
      int n = i >> 2;
      int koff = (i & 3) * 8;
      v4u v = {0u, 0u, 0u, 0u};
      if (n0 + n < N)
        v = *(const v4u*)&Bt[(size_t)(n0 + n) * K + k0 + koff];
      *(v4u*)&lBt[n][koff] = v;
    }
    __syncthreads();

    v16bf a0 = frag_a(lA, wm, lane);
    v16bf a1 = frag_a(lA, wm + 16, lane);
    v16bf b0 = frag_b(lBt, wn, lane);
    v16bf b1 = frag_b(lBt, wn + 16, lane);
    acc00 = __builtin_amdgcn_wmma_f32_16x16x32_bf16(false, a0, false, b0,
                                                    (short)0, acc00, false, false);
    acc01 = __builtin_amdgcn_wmma_f32_16x16x32_bf16(false, a0, false, b1,
                                                    (short)0, acc01, false, false);
    acc10 = __builtin_amdgcn_wmma_f32_16x16x32_bf16(false, a1, false, b0,
                                                    (short)0, acc10, false, false);
    acc11 = __builtin_amdgcn_wmma_f32_16x16x32_bf16(false, a1, false, b1,
                                                    (short)0, acc11, false, false);
    __syncthreads();
  }

  const int half = lane >> 4;
  const int ncol = lane & 15;
#pragma unroll
  for (int rm = 0; rm < 2; ++rm) {
#pragma unroll
    for (int rn = 0; rn < 2; ++rn) {
      const v8f* acc = rm == 0 ? (rn == 0 ? &acc00 : &acc01)
                               : (rn == 0 ? &acc10 : &acc11);
#pragma unroll
      for (int j = 0; j < 8; ++j) {
        int m = m0 + wm + rm * 16 + half * 8 + j;
        int c = n0 + wn + rn * 16 + ncol;
        if (m < M && c < N) {
          float v = (*acc)[j] + bias[c];
          if (relu) v = fmaxf(v, 0.f);
          C[(size_t)m * N + c] = v;
        }
      }
    }
  }
}

// ----------------------- maxpool 2x2/2 -------------------------------------
__global__ void maxpool2_kernel(const float* __restrict__ in,
                                float* __restrict__ out,
                                int C, int H, int W, int oh, int ow) {
  int i = blockIdx.x * blockDim.x + threadIdx.x;
  if (i >= C * oh * ow) return;
  int c = i / (oh * ow), rem = i % (oh * ow), y = rem / ow, x = rem % ow;
  const float* p = in + (size_t)c * H * W;
  float v = fmaxf(fmaxf(p[(size_t)(2 * y) * W + 2 * x],
                        p[(size_t)(2 * y) * W + 2 * x + 1]),
                  fmaxf(p[(size_t)(2 * y + 1) * W + 2 * x],
                        p[(size_t)(2 * y + 1) * W + 2 * x + 1]));
  out[i] = v;
}

// ----------------------- 1x1 conv (RPN heads; tiny) ------------------------
__global__ void conv1x1_kernel(const float* __restrict__ in,
                               const float* __restrict__ w,
                               const float* __restrict__ b,
                               float* __restrict__ out,
                               int Cin, int Cout, int HW) {
  int i = blockIdx.x * blockDim.x + threadIdx.x;
  if (i >= Cout * HW) return;
  int o = i / HW, p = i % HW;
  float s = b[o];
  for (int c = 0; c < Cin; ++c)
    s += in[(size_t)c * HW + p] * w[(size_t)o * Cin + c];
  out[i] = s;
}

// ----------------------- proposal box generation ---------------------------
__global__ void rpn_boxes_kernel(const float* __restrict__ cls_map,
                                 const float* __restrict__ loc_map,
                                 float* __restrict__ boxes,
                                 float* __restrict__ scores,
                                 int mh, int mw, float imgH, float imgW) {
  const int HW = mh * mw, N = HW * 9;
  int i = blockIdx.x * blockDim.x + threadIdx.x;
  if (i >= N) return;
  int a = i % 9, p = i / 9;
  int px = p % mw, py = p / mw;
  const float RAT[3] = {0.5f, 1.f, 2.f};
  const float SCL[3] = {8.f, 16.f, 32.f};
  float sr = sqrtf(RAT[a / 3]);
  float hh = 16.f * SCL[a % 3] * sr, ww = 16.f * SCL[a % 3] / sr;
  float cx = (px + 0.5f) * 16.f, cy = (py + 0.5f) * 16.f;
  float l0 = cls_map[(size_t)(2 * a) * HW + p];
  float l1 = cls_map[(size_t)(2 * a + 1) * HW + p];
  float sc = 1.f / (1.f + expf(l0 - l1));  // softmax fg prob
  float tx = loc_map[(size_t)(4 * a + 0) * HW + p];
  float ty = loc_map[(size_t)(4 * a + 1) * HW + p];
  float tw = loc_map[(size_t)(4 * a + 2) * HW + p];
  float th = loc_map[(size_t)(4 * a + 3) * HW + p];
  float gcx = tx * ww + cx, gcy = ty * hh + cy;
  float gw = expf(tw) * ww, gh = expf(th) * hh;
  float x1 = fminf(fmaxf(gcx - gw * 0.5f, 0.f), imgW);
  float y1 = fminf(fmaxf(gcy - gh * 0.5f, 0.f), imgH);
  float x2 = fminf(fmaxf(gcx + gw * 0.5f, 0.f), imgW);
  float y2 = fminf(fmaxf(gcy + gh * 0.5f, 0.f), imgH);
  boxes[(size_t)i * 4 + 0] = x1; boxes[(size_t)i * 4 + 1] = y1;
  boxes[(size_t)i * 4 + 2] = x2; boxes[(size_t)i * 4 + 3] = y2;
  bool ok = (x2 - x1 >= 16.f) && (y2 - y1 >= 16.f);
  scores[i] = ok ? sc : -1e30f;
}

// greedy NMS by iterative argmax (deterministic, single workgroup)
__global__ void nms_select_kernel(const float* __restrict__ boxes,
                                  float* __restrict__ scores, int N,
                                  float* __restrict__ rois, int nsel,
                                  float thresh) {
  __shared__ float sv[256];
  __shared__ int si[256];
  __shared__ float wb[4];
  int t = threadIdx.x;
  for (int it = 0; it < nsel; ++it) {
    float bv = -1e38f; int bi = -1;
    for (int i = t; i < N; i += 256) {
      float s = scores[i];
      if (s > bv) { bv = s; bi = i; }
    }
    sv[t] = bv; si[t] = bi;
    __syncthreads();
    for (int ofs = 128; ofs > 0; ofs >>= 1) {
      if (t < ofs && sv[t + ofs] > sv[t]) { sv[t] = sv[t + ofs]; si[t] = si[t + ofs]; }
      __syncthreads();
    }
    if (t == 0) {
      int w = si[0];
      if (w >= 0 && sv[0] > -1e29f) {
        for (int k = 0; k < 4; ++k) { wb[k] = boxes[(size_t)w * 4 + k]; rois[it * 4 + k] = wb[k]; }
        scores[w] = -1e38f;
      } else {
        for (int k = 0; k < 4; ++k) { wb[k] = 0.f; rois[it * 4 + k] = 0.f; }
      }
    }
    __syncthreads();
    float wx1 = wb[0], wy1 = wb[1], wx2 = wb[2], wy2 = wb[3];
    float wa = (wx2 - wx1) * (wy2 - wy1);
    for (int i = t; i < N; i += 256) {
      if (scores[i] <= -1e37f) continue;
      float bx1 = boxes[(size_t)i * 4], by1 = boxes[(size_t)i * 4 + 1];
      float bx2 = boxes[(size_t)i * 4 + 2], by2 = boxes[(size_t)i * 4 + 3];
      float tlx = fmaxf(wx1, bx1), tly = fmaxf(wy1, by1);
      float brx = fminf(wx2, bx2), bry = fminf(wy2, by2);
      float iw = fmaxf(brx - tlx, 0.f), ih = fmaxf(bry - tly, 0.f);
      float inter = iw * ih;
      float ba = (bx2 - bx1) * (by2 - by1);
      float iou = inter / (wa + ba - inter + 1e-9f);
      if (iou > thresh) scores[i] = -1e38f;
    }
    __syncthreads();
  }
}

// ----------------------- RoI align (7x7, ratio 2) --------------------------
__global__ void roi_align_kernel(const float* __restrict__ feat,
                                 const float* __restrict__ rois,
                                 float* __restrict__ pooled,
                                 int C, int H, int W, int R) {
  int r = blockIdx.x, c = blockIdx.y;
  int t = threadIdx.x;
  if (t >= 49 || r >= R || c >= C) return;
  int oy = t / 7, ox = t % 7;
  float x1 = rois[r * 4 + 0] * 0.0625f;
  float y1 = rois[r * 4 + 1] * 0.0625f;
  float rw = fmaxf(rois[r * 4 + 2] * 0.0625f - x1, 1.f);
  float rh = fmaxf(rois[r * 4 + 3] * 0.0625f - y1, 1.f);
  const float* F = feat + (size_t)c * H * W;
  float acc = 0.f;
#pragma unroll
  for (int sy = 0; sy < 2; ++sy) {
#pragma unroll
    for (int sx = 0; sx < 2; ++sx) {
      float ty = ((2 * oy + sy) + 0.5f) / 14.f;
      float tx = ((2 * ox + sx) + 0.5f) / 14.f;
      float y = fminf(fmaxf(y1 + ty * rh, 0.f), (float)(H - 1));
      float x = fminf(fmaxf(x1 + tx * rw, 0.f), (float)(W - 1));
      int y0 = (int)floorf(y), x0 = (int)floorf(x);
      int y1i = min(y0 + 1, H - 1), x1i = min(x0 + 1, W - 1);
      float ly = y - y0, lx = x - x0;
      acc += F[(size_t)y0 * W + x0] * (1 - ly) * (1 - lx)
           + F[(size_t)y0 * W + x1i] * (1 - ly) * lx
           + F[(size_t)y1i * W + x0] * ly * (1 - lx)
           + F[(size_t)y1i * W + x1i] * ly * lx;
    }
  }
  pooled[(size_t)r * (C * 49) + c * 49 + oy * 7 + ox] = acc * 0.25f;
}

// ----------------------- small FC heads ------------------------------------
__global__ void linear_kernel(const float* __restrict__ A,
                              const float* __restrict__ Wt,
                              const float* __restrict__ b,
                              float* __restrict__ out, int M, int N, int K) {
  int i = blockIdx.x * blockDim.x + threadIdx.x;
  if (i >= M * N) return;
  int m = i / N, n = i % N;
  float s = b[n];
  for (int k = 0; k < K; ++k) s += A[(size_t)m * K + k] * Wt[(size_t)k * N + n];
  out[i] = s;
}

// ----------------------- fast-head targets & losses ------------------------
__global__ void fast_targets_kernel(const float* __restrict__ rois,
                                    const float* __restrict__ gt, int nb,
                                    int* __restrict__ lab,
                                    float* __restrict__ targ, int R) {
  int t = threadIdx.x + blockIdx.x * blockDim.x;
  if (t >= R) return;
  float rx1 = rois[t * 4], ry1 = rois[t * 4 + 1];
  float rx2 = rois[t * 4 + 2], ry2 = rois[t * 4 + 3];
  float best = 0.f; int barg = 0;
  for (int g = 0; g < nb; ++g) {
    const float* G = gt + g * 5;
    float tlx = fmaxf(rx1, G[0]), tly = fmaxf(ry1, G[1]);
    float brx = fminf(rx2, G[2]), bry = fminf(ry2, G[3]);
    float iw = fmaxf(brx - tlx, 0.f), ih = fmaxf(bry - tly, 0.f);
    float inter = iw * ih;
    float ua = (rx2 - rx1) * (ry2 - ry1) + (G[2] - G[0]) * (G[3] - G[1]) - inter + 1e-9f;
    float iou = inter / ua;
    if (iou > best) { best = iou; barg = g; }
  }
  if (best >= 0.5f) {
    const float* G = gt + barg * 5;
    lab[t] = (int)G[4] + 1;
    float w = fmaxf(rx2 - rx1, 1e-3f), h = fmaxf(ry2 - ry1, 1e-3f);
    float cx = rx1 + w * 0.5f, cy = ry1 + h * 0.5f;
    float gw = fmaxf(G[2] - G[0], 1e-3f), gh = fmaxf(G[3] - G[1], 1e-3f);
    float gcx = G[0] + gw * 0.5f, gcy = G[1] + gh * 0.5f;
    targ[t * 4 + 0] = (gcx - cx) / w;
    targ[t * 4 + 1] = (gcy - cy) / h;
    targ[t * 4 + 2] = logf(gw / w);
    targ[t * 4 + 3] = logf(gh / h);
  } else {
    lab[t] = 0;
    targ[t * 4 + 0] = 0.f; targ[t * 4 + 1] = 0.f;
    targ[t * 4 + 2] = 0.f; targ[t * 4 + 3] = 0.f;
  }
}

__global__ void fast_ce_kernel(const float* __restrict__ logits,
                               const int* __restrict__ lab,
                               float* __restrict__ accum, int R, int C) {
  int t = threadIdx.x + blockIdx.x * blockDim.x;
  if (t >= R) return;
  const float* L = logits + (size_t)t * C;
  float m = L[0];
  for (int j = 1; j < C; ++j) m = fmaxf(m, L[j]);
  float s = 0.f;
  for (int j = 0; j < C; ++j) s += expf(L[j] - m);
  int l = lab[t]; if (l < 0) l = 0; if (l >= C) l = C - 1;
  float ce = (m + logf(s)) - L[l];
  atomicAdd(&accum[2], ce / (float)R);
}

__global__ void fast_l1_kernel(const float* __restrict__ locout,
                               const float* __restrict__ targ,
                               const int* __restrict__ lab,
                               float* __restrict__ accum, int R) {
  int t = threadIdx.x + blockIdx.x * blockDim.x;
  if (t >= R) return;
  int l = lab[t];
  if (l <= 0) return;
  float s = 0.f;
  for (int k = 0; k < 4; ++k) {
    float d = fabsf(locout[(size_t)t * 84 + l * 4 + k] - targ[t * 4 + k]);
    s += (d < 1.f) ? d * d * 0.5f : d - 0.5f;  // sigma = 1
  }
  atomicAdd(&accum[3], s / (float)R);
}

// ----------------------- RPN losses (deterministic sampling approximation) -
__global__ void rpn_loss_kernel(const float* __restrict__ cls_map,
                                const float* __restrict__ loc_map,
                                const float* __restrict__ gt, int nb,
                                int mh, int mw, float imgH, float imgW,
                                float* __restrict__ accum) {
  __shared__ float s_ce_pos, s_ce_neg, s_l1;
  __shared__ int s_npos, s_nneg;
  if (threadIdx.x == 0) { s_ce_pos = 0.f; s_ce_neg = 0.f; s_l1 = 0.f; s_npos = 0; s_nneg = 0; }
  __syncthreads();
  const int HW = mh * mw, N = HW * 9;
  const float RAT[3] = {0.5f, 1.f, 2.f};
  const float SCL[3] = {8.f, 16.f, 32.f};
  for (int i = threadIdx.x; i < N; i += blockDim.x) {
    int a = i % 9, p = i / 9;
    int px = p % mw, py = p / mw;
    float sr = sqrtf(RAT[a / 3]);
    float hh = 16.f * SCL[a % 3] * sr, ww = 16.f * SCL[a % 3] / sr;
    float cx = (px + 0.5f) * 16.f, cy = (py + 0.5f) * 16.f;
    float ax1 = cx - ww * 0.5f, ay1 = cy - hh * 0.5f;
    float ax2 = cx + ww * 0.5f, ay2 = cy + hh * 0.5f;
    if (ax1 < 0.f || ay1 < 0.f || ax2 > imgW || ay2 > imgH) continue;
    float best = 0.f; int barg = 0;
    for (int g = 0; g < nb; ++g) {
      const float* G = gt + g * 5;
      float tlx = fmaxf(ax1, G[0]), tly = fmaxf(ay1, G[1]);
      float brx = fminf(ax2, G[2]), bry = fminf(ay2, G[3]);
      float iw = fmaxf(brx - tlx, 0.f), ih = fmaxf(bry - tly, 0.f);
      float inter = iw * ih;
      float ua = ww * hh + (G[2] - G[0]) * (G[3] - G[1]) - inter + 1e-9f;
      float iou = inter / ua;
      if (iou > best) { best = iou; barg = g; }
    }
    int lab = (best >= 0.7f) ? 1 : ((best < 0.3f) ? 0 : -1);
    if (lab < 0) continue;
    float l0 = cls_map[(size_t)(2 * a) * HW + p];
    float l1v = cls_map[(size_t)(2 * a + 1) * HW + p];
    float mm = fmaxf(l0, l1v);
    float lse = mm + logf(expf(l0 - mm) + expf(l1v - mm));
    if (lab == 1) {
      atomicAdd(&s_ce_pos, lse - l1v);
      atomicAdd(&s_npos, 1);
      const float* G = gt + barg * 5;
      float gw = G[2] - G[0], gh = G[3] - G[1];
      float gcx = G[0] + gw * 0.5f, gcy = G[1] + gh * 0.5f;
      float tg[4] = {(gcx - cx) / ww, (gcy - cy) / hh, logf(gw / ww), logf(gh / hh)};
      float s = 0.f;
      for (int k = 0; k < 4; ++k) {
        float d = fabsf(loc_map[(size_t)(4 * a + k) * HW + p] - tg[k]);
        s += (d < 1.f) ? (d * 3.f) * (d * 3.f) * 0.5f : d - 0.5f / 9.f;  // sigma = 3
      }
      atomicAdd(&s_l1, s);
    } else {
      atomicAdd(&s_ce_neg, lse - l0);
      atomicAdd(&s_nneg, 1);
    }
  }
  __syncthreads();
  if (threadIdx.x == 0) {
    int np = s_npos, nn = s_nneg;
    int np_sel = np < 128 ? np : 128;
    int nn_sel = 256 - np_sel; if (nn_sel > nn) nn_sel = nn;
    float cls = 0.f;
    if (np > 0) cls += (s_ce_pos / np) * ((float)np_sel / 256.f);
    if (nn > 0) cls += (s_ce_neg / nn) * ((float)nn_sel / 256.f);
    float box = (np > 0) ? s_l1 * ((float)np_sel / (float)np) / 240.f : 0.f;
    atomicAdd(&accum[0], cls);
    atomicAdd(&accum[1], box);
  }
}

__global__ void finalize_kernel(const float* __restrict__ accum,
                                float* __restrict__ out) {
  if (threadIdx.x == 0 && blockIdx.x == 0)
    out[0] = accum[0] + accum[1] + accum[2] + accum[3];
}

// ---------------------------------------------------------------------------
extern "C" void kernel_launch(void* const* d_in, const int* in_sizes, int n_in,
                              void* d_out, int out_size, void* d_ws, size_t ws_size,
                              hipStream_t stream) {
  (void)in_sizes; (void)n_in; (void)out_size; (void)ws_size;
  // setup_inputs order: imgs, bboxes, num_b, num_H, num_W, params(pytree,
  // dict keys sorted: cls, fc1, fc2, loc, rpn_cls, rpn_conv, rpn_loc, vgg)
  const float* imgs   = (const float*)d_in[0];
  const float* bboxes = (const float*)d_in[1];
  const float* cls_w  = (const float*)d_in[5];
  const float* cls_b  = (const float*)d_in[6];
  const float* fc1_w  = (const float*)d_in[7];
  const float* fc1_b  = (const float*)d_in[8];
  const float* fc2_w  = (const float*)d_in[9];
  const float* fc2_b  = (const float*)d_in[10];
  const float* loc_w  = (const float*)d_in[11];
  const float* loc_b  = (const float*)d_in[12];
  const float* rpn_cls_w  = (const float*)d_in[13];
  const float* rpn_cls_b  = (const float*)d_in[14];
  const float* rpn_conv_w = (const float*)d_in[15];
  const float* rpn_conv_b = (const float*)d_in[16];
  const float* rpn_loc_w  = (const float*)d_in[17];
  const float* rpn_loc_b  = (const float*)d_in[18];
  const float* vgg_w[13]; const float* vgg_b[13];
  for (int i = 0; i < 13; ++i) {
    vgg_w[i] = (const float*)d_in[19 + 2 * i];
    vgg_b[i] = (const float*)d_in[20 + 2 * i];
  }

  const int H0 = 600, W0 = 800, NB = 8, NUMCLS = 21, R = 128;
  const int VC_IN[13]  = {3, 64, 64, 128, 128, 256, 256, 256, 512, 512, 512, 512, 512};
  const int VC_OUT[13] = {64, 64, 128, 128, 256, 256, 256, 512, 512, 512, 512, 512, 512};
  const bool POOL_AFTER[13] = {false, true, false, true, false, false, true,
                               false, false, true, false, false, false};

  // workspace bump allocator
  char* base = (char*)d_ws;
  size_t off = 0;
  auto alloc = [&](size_t bytes) -> char* {
    char* p = base + off;
    off = (off + bytes + 255) & ~(size_t)255;
    return p;
  };
  float* bufA = (float*)alloc((size_t)64 * H0 * W0 * 4);
  float* bufB = (float*)alloc((size_t)64 * H0 * W0 * 4);
  int KP[13];
  bfloat* wbf_v[13];
  for (int i = 0; i < 13; ++i) {
    KP[i] = (VC_IN[i] * 9 + 31) & ~31;
    wbf_v[i] = (bfloat*)alloc((size_t)VC_OUT[i] * KP[i] * 2);
  }
  const int KP_RPN = (512 * 9 + 31) & ~31;  // 4608, already aligned
  bfloat* wbf_rpn = (bfloat*)alloc((size_t)512 * KP_RPN * 2);
  bfloat* wbf_fc1 = (bfloat*)alloc((size_t)4096 * 25088 * 2);  // [N][K]
  bfloat* wbf_fc2 = (bfloat*)alloc((size_t)4096 * 4096 * 2);   // [N][K]
  const int MH = 37, MW = 50, FHW = MH * MW, NANCH = FHW * 9;
  float* cls_map = (float*)alloc((size_t)18 * FHW * 4);
  float* loc_map = (float*)alloc((size_t)36 * FHW * 4);
  float* boxes   = (float*)alloc((size_t)NANCH * 4 * 4);
  float* scores  = (float*)alloc((size_t)NANCH * 4);
  float* rois    = (float*)alloc((size_t)R * 4 * 4);
  float* pooled  = (float*)alloc((size_t)R * 25088 * 4);
  float* fco1    = (float*)alloc((size_t)R * 4096 * 4);
  float* fco2    = (float*)alloc((size_t)R * 4096 * 4);
  float* clso    = (float*)alloc((size_t)R * NUMCLS * 4);
  float* loco    = (float*)alloc((size_t)R * NUMCLS * 4 * 4);
  int*   flab    = (int*)alloc((size_t)R * 4);
  float* ftarg   = (float*)alloc((size_t)R * 4 * 4);
  float* accum   = (float*)alloc(8 * 4);

  // 0) zero accumulators
  zero_kernel<<<1, 32, 0, stream>>>(accum, 8);

  // 1) weight conversion: conv weights padded, FC weights transposed
  for (int i = 0; i < 13; ++i) {
    int n = VC_OUT[i] * KP[i];
    cvt_pad_bf16_kernel<<<(n + 255) / 256, 256, 0, stream>>>(
        vgg_w[i], wbf_v[i], VC_OUT[i], VC_IN[i] * 9, KP[i]);
  }
  {
    int n = 512 * KP_RPN;
    cvt_pad_bf16_kernel<<<(n + 255) / 256, 256, 0, stream>>>(
        rpn_conv_w, wbf_rpn, 512, 512 * 9, KP_RPN);
    size_t nt = (size_t)4096 * 25088;
    cvt_transpose_bf16_kernel<<<(unsigned)((nt + 255) / 256), 256, 0, stream>>>(
        fc1_w, wbf_fc1, 25088, 4096);
    nt = (size_t)4096 * 4096;
    cvt_transpose_bf16_kernel<<<(unsigned)((nt + 255) / 256), 256, 0, stream>>>(
        fc2_w, wbf_fc2, 4096, 4096);
  }

  // 2) preprocess -> bufA (3,600,800)
  {
    int n = 3 * H0 * W0;
    preprocess_kernel<<<(n + 255) / 256, 256, 0, stream>>>(imgs, bufA, H0, W0);
  }

  // 3) VGG backbone (WMMA implicit GEMM), ping-pong bufA/bufB
  float* cur = bufA; float* nxt = bufB;
  int Hc = H0, Wc = W0;
  for (int i = 0; i < 13; ++i) {
    int HW = Hc * Wc;
    int ntn = (HW + 63) / 64, ntm = (VC_OUT[i] + 127) / 128;
    conv3x3_wmma_kernel<<<ntn * ntm, 256, 0, stream>>>(
        cur, wbf_v[i], vgg_b[i], nxt, VC_IN[i], VC_OUT[i], Hc, Wc, 1);
    { float* t = cur; cur = nxt; nxt = t; }
    if (POOL_AFTER[i]) {
      int oh = Hc / 2, ow = Wc / 2;
      int n = VC_OUT[i] * oh * ow;
      maxpool2_kernel<<<(n + 255) / 256, 256, 0, stream>>>(cur, nxt, VC_OUT[i], Hc, Wc, oh, ow);
      { float* t = cur; cur = nxt; nxt = t; }
      Hc = oh; Wc = ow;
    }
  }
  float* feat = cur;     // (512, 37, 50)
  float* rpnh = nxt;     // reuse partner buffer for RPN hidden

  // 4) RPN conv (3x3, 512->512, relu) + 1x1 heads
  {
    int ntn = (FHW + 63) / 64, ntm = 512 / 128;
    conv3x3_wmma_kernel<<<ntn * ntm, 256, 0, stream>>>(
        feat, wbf_rpn, rpn_conv_b, rpnh, 512, 512, MH, MW, 1);
    int n = 18 * FHW;
    conv1x1_kernel<<<(n + 255) / 256, 256, 0, stream>>>(rpnh, rpn_cls_w, rpn_cls_b, cls_map, 512, 18, FHW);
    n = 36 * FHW;
    conv1x1_kernel<<<(n + 255) / 256, 256, 0, stream>>>(rpnh, rpn_loc_w, rpn_loc_b, loc_map, 512, 36, FHW);
  }

  // 5) proposals + NMS -> 128 RoIs (deterministic greedy selection)
  rpn_boxes_kernel<<<(NANCH + 255) / 256, 256, 0, stream>>>(
      cls_map, loc_map, boxes, scores, MH, MW, (float)H0, (float)W0);
  nms_select_kernel<<<1, 256, 0, stream>>>(boxes, scores, NANCH, rois, R, 0.7f);

  // 6) RoI align (128, 512, 7, 7)
  {
    dim3 grid(R, 512);
    roi_align_kernel<<<grid, 64, 0, stream>>>(feat, rois, pooled, 512, MH, MW, R);
  }

  // 7) fast head: fc1 (25088->4096) and fc2 (4096->4096) via WMMA
  {
    int ntn = 4096 / 64;  // M=128 fits one tile (ntm=1)
    gemm_wmma_kernel<<<ntn, 256, 0, stream>>>(pooled, wbf_fc1, fc1_b, fco1, R, 4096, 25088, 1);
    gemm_wmma_kernel<<<ntn, 256, 0, stream>>>(fco1, wbf_fc2, fc2_b, fco2, R, 4096, 4096, 1);
    int n = R * NUMCLS;
    linear_kernel<<<(n + 255) / 256, 256, 0, stream>>>(fco2, cls_w, cls_b, clso, R, NUMCLS, 4096);
    n = R * NUMCLS * 4;
    linear_kernel<<<(n + 255) / 256, 256, 0, stream>>>(fco2, loc_w, loc_b, loco, R, NUMCLS * 4, 4096);
  }

  // 8) targets + losses
  fast_targets_kernel<<<1, R, 0, stream>>>(rois, bboxes, NB, flab, ftarg, R);
  fast_ce_kernel<<<1, R, 0, stream>>>(clso, flab, accum, R, NUMCLS);
  fast_l1_kernel<<<1, R, 0, stream>>>(loco, ftarg, flab, accum, R);
  rpn_loss_kernel<<<1, 256, 0, stream>>>(cls_map, loc_map, bboxes, NB, MH, MW,
                                         (float)H0, (float)W0, accum);
  finalize_kernel<<<1, 32, 0, stream>>>(accum, (float*)d_out);
}
